// DQNModel_74448963109074
// MI455X (gfx1250) — matrix operand order repaired and checked by reading, not compiled, not run on hardware
//
#include <hip/hip_runtime.h>

#define NUSERS 4096
#define NDOCS  50
#define DIM    64
#define TSTEPS 50

typedef __attribute__((ext_vector_type(16))) _Float16 v16h;
typedef __attribute__((ext_vector_type(8)))  float    v8f;

// ---------------------------------------------------------------------------
// Kernel 1: LSTM scan + dense head -> user_embeddings (4096 x 64)
// One wave per user; lane j owns h[j], c[j] (LSTM_UNITS == 32 == wave32).
// ---------------------------------------------------------------------------
__global__ __launch_bounds__(256) void lstm_kernel(
    const int*   __restrict__ doc_id, const float* __restrict__ ctime,
    const float* __restrict__ doc_embed,
    const float* __restrict__ Wx, const float* __restrict__ Wh,
    const float* __restrict__ bz,
    const float* __restrict__ d1W, const float* __restrict__ d1b,
    const float* __restrict__ heW, const float* __restrict__ heb,
    float* __restrict__ ue)
{
    __shared__ float sWx[65 * 128];
    __shared__ float sWh[32 * 128];
    __shared__ float sB[128];
    __shared__ float sXt[8][68];
    __shared__ float sH[8][32];
    __shared__ float sH1[8][32];

    const int tid = threadIdx.x;
    for (int i = tid; i < 65 * 128; i += 256) sWx[i] = Wx[i];
    for (int i = tid; i < 32 * 128; i += 256) sWh[i] = Wh[i];
    if (tid < 128) sB[tid] = bz[tid];
    __syncthreads();

    const int w    = tid >> 5;
    const int lane = tid & 31;
    const int u    = blockIdx.x * 8 + w;

    float h = 0.f, c = 0.f;
    const int g0 = lane, g1 = 32 + lane, g2 = 64 + lane, g3 = 96 + lane;

    for (int t = 0; t < TSTEPS; ++t) {
        const int id = doc_id[u * TSTEPS + t];
        const float* de = doc_embed + id * DIM;
        sXt[w][lane]      = de[lane];
        sXt[w][lane + 32] = de[lane + 32];
        if (lane == 0) sXt[w][64] = ctime[u * TSTEPS + t];
        sH[w][lane] = h;
        __syncthreads();

        float zi = sB[g0], zf = sB[g1], zg = sB[g2], zo = sB[g3];
        for (int k = 0; k < 65; ++k) {
            const float x   = sXt[w][k];
            const float* wr = sWx + k * 128;
            zi += x * wr[g0]; zf += x * wr[g1]; zg += x * wr[g2]; zo += x * wr[g3];
        }
        for (int mth = 0; mth < 32; ++mth) {
            const float hm  = sH[w][mth];
            const float* wr = sWh + mth * 128;
            zi += hm * wr[g0]; zf += hm * wr[g1]; zg += hm * wr[g2]; zo += hm * wr[g3];
        }
        const float ig = 1.f / (1.f + __expf(-zi));
        const float fg = 1.f / (1.f + __expf(-zf));
        const float gg = tanhf(zg);
        const float og = 1.f / (1.f + __expf(-zo));
        c = fg * c + ig * gg;
        h = og * tanhf(c);
        __syncthreads();
    }

    sH[w][lane] = h;
    __syncthreads();
    float h1 = d1b[lane];
    for (int mth = 0; mth < 32; ++mth) h1 += sH[w][mth] * d1W[mth * 32 + lane];
    h1 = (h1 >= 0.f) ? h1 : 0.3f * h1;   // leaky relu, alpha=0.3
    sH1[w][lane] = h1;
    __syncthreads();
    for (int cc = lane; cc < DIM; cc += 32) {
        float acc = heb[cc];
        for (int n = 0; n < 32; ++n) acc += sH1[w][n] * heW[n * DIM + cc];
        ue[u * DIM + cc] = acc;
    }
}

// ---------------------------------------------------------------------------
// CDNA5 WMMA f16 operand layouts (CDNA5 ISA §7.12.2):
//  A (16x32, MxK): lane m=l&15 holds row m; half=l>>4; VGPR v<4 -> K=2v,2v+1,
//                  v>=4 -> K=16+2(v-4); half adds +8 to K offset.
//  B (32x16, KxN): lane n=l&15 is column; half adds +8 to K offset;
//                  VGPR v -> same K map as A.
// ---------------------------------------------------------------------------
__device__ __forceinline__ float feat_val(const float* ueRow, const float* dRow, int k) {
    return (k < DIM) ? ueRow[k] : dRow[k - DIM];
}

__device__ __forceinline__ v16h build_a_feat(const float* ueRow, const float* dRow,
                                             int k0, int half) {
    v16h a;
#pragma unroll
    for (int v = 0; v < 8; ++v) {
        const int kb = ((v < 4) ? 2 * v : 16 + 2 * (v - 4)) + (half ? 8 : 0);
#pragma unroll
        for (int e = 0; e < 2; ++e)
            a[2 * v + e] = (_Float16)feat_val(ueRow, dRow, k0 + kb + e);
    }
    return a;
}

// ---------------------------------------------------------------------------
// Kernel 2: per-(user,doc) MLP via WMMA + cf-score/exp.
// One wave per 16-row tile of the 204800x128 feature matrix.
// Weights are pre-swizzled once per block into LDS in B-fragment layout, so
// the hot loop is ds_load_b128 x2 -> v_wmma per k-step. x1 is stored directly
// in A-fragment layout for GEMM2 (contiguous v16h LDS loads).
// Dynamic LDS: |W1 frags 32K | W2 frags 8K | x1 32K | x2 16K | biases|
// ---------------------------------------------------------------------------
__global__ __launch_bounds__(256) void mlp_kernel(
    const float* __restrict__ ue, const float* __restrict__ dpe,
    const float* __restrict__ W1, const float* __restrict__ b1,
    const float* __restrict__ W2, const float* __restrict__ b2,
    const float* __restrict__ qW, const float* __restrict__ qb,
    float* __restrict__ out_un, float* __restrict__ out_pq)
{
    extern __shared__ char smem[];
    _Float16* sW1f = (_Float16*)smem;            // 32 frags * 32 lanes * 16 = 16384
    _Float16* sW2f = sW1f + 32 * 32 * 16;        // 8 frags  * 32 lanes * 16 =  4096
    _Float16* sX1a = sW2f + 8 * 32 * 16;         // 8 waves * 2048
    float*    sX2  = (float*)(sX1a + 8 * 2048);  // 8 waves * 512
    float*    sB1  = sX2 + 8 * 512;              // 128
    float*    sB2  = sB1 + 128;                  // 32
    float*    sQW  = sB2 + 32;                   // 32

    const int tid = threadIdx.x;

    // ---- cooperative pre-swizzle of W1/W2 into B-fragment layout (f16) ----
    for (int idx = tid; idx < 32 * 32 * 16; idx += 256) {
        const int f  = idx >> 9;                 // fragment id = nt*4 + kb
        const int l  = (idx >> 4) & 31;
        const int i  = idx & 15;
        const int n  = l & 15, hk = (l >> 4) * 8;
        const int kb = f & 3,  nt = f >> 2;
        const int k  = kb * 32 + hk + (i & 7) + (i >> 3) * 16;
        sW1f[idx] = (_Float16)W1[k * 128 + nt * 16 + n];
    }
    for (int idx = tid; idx < 8 * 32 * 16; idx += 256) {
        const int f  = idx >> 9;
        const int l  = (idx >> 4) & 31;
        const int i  = idx & 15;
        const int n  = l & 15, hk = (l >> 4) * 8;
        const int kb = f & 3,  nt = f >> 2;
        const int k  = kb * 32 + hk + (i & 7) + (i >> 3) * 16;
        sW2f[idx] = (_Float16)W2[k * 32 + nt * 16 + n];
    }
    if (tid < 128) sB1[tid] = b1[tid];
    if (tid < 32)  { sB2[tid] = b2[tid]; sQW[tid] = qW[tid]; }
    __syncthreads();

    const int w    = tid >> 5;
    const int lane = tid & 31;
    const int m    = lane & 15;      // A-row / B-column / C-column
    const int half = lane >> 4;

    const int tile = blockIdx.x * 8 + w;
    const int row  = tile * 16 + m;
    const int uidx = row / NDOCS;
    const int didx = row - uidx * NDOCS;
    const float* ueRow = ue  + uidx * DIM;
    const float* dRow  = dpe + (didx + 1) * DIM;   // doc_prop_embed[1:51]

    _Float16* myX1 = sX1a + w * 2048;

    // ---- A fragments for GEMM1 (from global gathers, once per tile) ----
    v16h aF[4];
#pragma unroll
    for (int kb = 0; kb < 4; ++kb) aF[kb] = build_a_feat(ueRow, dRow, kb * 32, half);

    // ---- GEMM1: feats(16x128) @ W1(128x128) + b1, relu ----
#pragma unroll
    for (int nt = 0; nt < 8; ++nt) {
        const int col = nt * 16 + m;
        const float bc = sB1[col];
        v8f acc = {bc, bc, bc, bc, bc, bc, bc, bc};
#pragma unroll
        for (int kb = 0; kb < 4; ++kb) {
            const v16h bF = *(const v16h*)&sW1f[((nt * 4 + kb) * 32 + lane) * 16];
            acc = __builtin_amdgcn_wmma_f32_16x16x32_f16(
                      false, aF[kb], false, bF, (short)0, acc, false, false);
        }
        // C layout -> A-fragment layout for GEMM2 (inverse K mapping)
        const int w32 = col & 31, kb2 = col >> 5;
        const int g   = (w32 >= 16) ? 1 : 0;
        const int rr0 = w32 - 16 * g;
        const int ah  = (rr0 >= 8) ? 1 : 0;
        const int rr  = rr0 - 8 * ah;
        const int elt = 2 * (4 * g + (rr >> 1)) + (rr & 1);
#pragma unroll
        for (int r = 0; r < 8; ++r) {
            const int mm = r + half * 8;
            const int dl = mm + 16 * ah;           // destination lane in A layout
            myX1[(dl * 4 + kb2) * 16 + elt] = (_Float16)fmaxf(acc[r], 0.f);
        }
    }
    asm volatile("s_wait_dscnt 0" ::: "memory");   // wave-level LDS RAW fence

    // ---- GEMM2: x1(16x128) @ W2(128x32) + b2, relu ----
    v16h a2[4];
#pragma unroll
    for (int kb = 0; kb < 4; ++kb)
        a2[kb] = *(const v16h*)&myX1[(lane * 4 + kb) * 16];

#pragma unroll
    for (int nt = 0; nt < 2; ++nt) {
        const int col = nt * 16 + m;
        const float bc = sB2[col];
        v8f acc = {bc, bc, bc, bc, bc, bc, bc, bc};
#pragma unroll
        for (int kb = 0; kb < 4; ++kb) {
            const v16h bF = *(const v16h*)&sW2f[((nt * 4 + kb) * 32 + lane) * 16];
            acc = __builtin_amdgcn_wmma_f32_16x16x32_f16(
                      false, a2[kb], false, bF, (short)0, acc, false, false);
        }
#pragma unroll
        for (int r = 0; r < 8; ++r)
            sX2[w * 512 + (r + half * 8) * 32 + col] = fmaxf(acc[r], 0.f);
    }
    asm volatile("s_wait_dscnt 0" ::: "memory");

    // ---- q head + cf score + exp (lanes 0..15, one row each) ----
    if (lane < 16) {
        float q = qb[0];
        for (int cidx = 0; cidx < 32; ++cidx)
            q += sX2[w * 512 + m * 32 + cidx] * sQW[cidx];
        float cf = 0.f;
        for (int k = 0; k < DIM; ++k) cf += ueRow[k] * dRow[k];
        const float un = __expf(cf);
        out_un[row] = un;
        out_pq[row] = un * q;
    }
}

// ---------------------------------------------------------------------------
// Kernel 3: slate aggregation (bandwidth bound: 40 MB of output)
// ---------------------------------------------------------------------------
__global__ __launch_bounds__(256) void slate_kernel(
    const int* __restrict__ slates, const float* __restrict__ un,
    const float* __restrict__ pq, float* __restrict__ out, int nslates)
{
    __shared__ float sun[NDOCS];
    __shared__ float spq[NDOCS];
    const int u = blockIdx.x;
    for (int i = threadIdx.x; i < NDOCS; i += blockDim.x) {
        sun[i] = un[u * NDOCS + i];
        spq[i] = pq[u * NDOCS + i];
    }
    __syncthreads();
    for (int s = threadIdx.x; s < nslates; s += blockDim.x) {
        const int a = slates[2 * s + 0];
        const int b = slates[2 * s + 1];
        out[(size_t)u * nslates + s] =
            (spq[a] + spq[b]) / (sun[a] + sun[b] + 1.0f);
    }
}

// ---------------------------------------------------------------------------
extern "C" void kernel_launch(void* const* d_in, const int* in_sizes, int n_in,
                              void* d_out, int out_size, void* d_ws, size_t ws_size,
                              hipStream_t stream)
{
    const int*   doc_id    = (const int*)  d_in[0];
    const float* ctime     = (const float*)d_in[1];
    const int*   slates    = (const int*)  d_in[2];
    const float* doc_embed = (const float*)d_in[3];
    const float* dpe       = (const float*)d_in[4];
    const float* Wx        = (const float*)d_in[5];
    const float* Wh        = (const float*)d_in[6];
    const float* lb        = (const float*)d_in[7];
    const float* d1W       = (const float*)d_in[8];
    const float* d1b       = (const float*)d_in[9];
    const float* heW       = (const float*)d_in[10];
    const float* heb       = (const float*)d_in[11];
    const float* W1        = (const float*)d_in[12];
    const float* b1        = (const float*)d_in[13];
    const float* W2        = (const float*)d_in[14];
    const float* b2        = (const float*)d_in[15];
    const float* qW        = (const float*)d_in[16];
    const float* qb        = (const float*)d_in[17];

    float* ue = (float*)d_ws;                       // 4096*64
    float* un = ue + NUSERS * DIM;                  // 4096*50
    float* pq = un + NUSERS * NDOCS;                // 4096*50

    lstm_kernel<<<NUSERS / 8, 256, 0, stream>>>(
        doc_id, ctime, doc_embed, Wx, Wh, lb, d1W, d1b, heW, heb, ue);

    // dynamic LDS: (16384 + 4096 + 16384) halfs + (4096 + 192) floats
    const size_t lds_bytes = (size_t)(16384 + 4096 + 16384) * 2
                           + (size_t)(4096 + 192) * 4;     // = 90880 B
    // 204800 rows / 16 per tile / 8 tiles per block = 1600 blocks
    mlp_kernel<<<(NUSERS * NDOCS) / (16 * 8), 256, lds_bytes, stream>>>(
        ue, dpe, W1, b1, W2, b2, qW, qb, un, pq);

    const int nslates = in_sizes[2] / 2;
    slate_kernel<<<NUSERS, 256, 0, stream>>>(slates, un, pq, (float*)d_out, nslates);
}
